// AttentionBlock_55920474194459
// MI455X (gfx1250) — matrix operand order repaired and checked by reading, not compiled, not run on hardware
//
#include <hip/hip_runtime.h>
#include <hip/hip_bf16.h>

typedef _Float16 half_t;
typedef __attribute__((ext_vector_type(16))) _Float16 v16h;
typedef __attribute__((ext_vector_type(8)))  _Float16 v8h;
typedef __attribute__((ext_vector_type(8)))  float    v8f;

union FragH { v16h v; v8h h[2]; };

// A-fragment (16x32 f16): p points at rowbase + koff (koff = lane>=16 ? 16 : 0),
// 16 contiguous halves K koff..koff+15.
__device__ __forceinline__ v16h ld_frag_a(const half_t* p) {
    FragH f;
    f.h[0] = *(const v8h*)(p);
    f.h[1] = *(const v8h*)(p + 8);
    return f.v;
}
// B-fragment (32x16 f16) for A*B^T with B stored row-major [n][k]:
// 16 contiguous halves at row (lane&15), kbase=(lane>=16)?16:0.
__device__ __forceinline__ v16h ld_frag_b(const half_t* p) {
    FragH f;
    f.h[0] = *(const v8h*)(p);
    f.h[1] = *(const v8h*)(p + 8);
    return f.v;
}

__device__ __forceinline__ v8f wmma_f16(v16h a, v16h b, v8f c) {
    return __builtin_amdgcn_wmma_f32_16x16x32_f16(false, a, false, b, (short)0, c, false, false);
}

// CDNA5 async copy, GVS mode: SGPR 64-bit base + 32-bit VGPR byte offset.
// 16B per lane -> LDS, tracked by ASYNCcnt.
__device__ __forceinline__ void async_ld16(void* lds_ptr, const half_t* sbase, unsigned voff) {
    unsigned lds = (unsigned)(uintptr_t)lds_ptr;   // addr[31:0] == LDS byte address
    asm volatile("global_load_async_to_lds_b128 %0, %1, %2"
                 :: "v"(lds), "v"(voff), "s"(sbase) : "memory");
}
#define ASYNC_WAIT() asm volatile("s_wait_asynccnt 0x0" ::: "memory")
#define WAVE_LDS_FENCE() asm volatile("" ::: "memory")  // same-wave LDS is HW in-order

#define N_TOK   4096
#define C_DIM   256
#define B_DIM   4
#define LOG2E   1.44269504088896340736f

// ---------------------------------------------------------------- weights f32->f16
__global__ __launch_bounds__(256) void cvt_weights(const float* __restrict__ qkvw,
                                                   const float* __restrict__ projw,
                                                   half_t* __restrict__ qkvw_h,
                                                   half_t* __restrict__ projw_h) {
    int i = blockIdx.x * 256 + threadIdx.x;
    if (i < 3 * C_DIM * C_DIM) qkvw_h[i] = (half_t)qkvw[i];
    if (i < C_DIM * C_DIM)     projw_h[i] = (half_t)projw[i];
}

// ---------------------------------------------------------------- GroupNorm -> f16 tokens [b,n,c]
__global__ __launch_bounds__(256) void gn_kernel(const float* __restrict__ x,
                                                 const float* __restrict__ gw,
                                                 const float* __restrict__ gb,
                                                 half_t* __restrict__ tokens) {
    const int tid = threadIdx.x;
    const int bi = blockIdx.x >> 3, g = blockIdx.x & 7;     // 8 groups of 32 ch
    const float* base = x + (bi * C_DIM + g * 32) * N_TOK;
    const float4* b4 = (const float4*)base;
    const int GE4 = 32 * N_TOK / 4;                          // 32768 float4

    float s = 0.f, ss = 0.f;
    for (int i = tid; i < GE4; i += 256) {
        float4 v = b4[i];
        s  += v.x + v.y + v.z + v.w;
        ss += v.x * v.x + v.y * v.y + v.z * v.z + v.w * v.w;
    }
    __shared__ float sh1[256], sh2[256];
    sh1[tid] = s; sh2[tid] = ss;
    __syncthreads();
    for (int st = 128; st > 0; st >>= 1) {
        if (tid < st) { sh1[tid] += sh1[tid + st]; sh2[tid] += sh2[tid + st]; }
        __syncthreads();
    }
    const float inv = 1.0f / (float)(32 * N_TOK);
    const float mean = sh1[0] * inv;
    const float var  = sh2[0] * inv - mean * mean;
    const float rstd = rsqrtf(var + 1e-5f);

    for (int i = tid; i < GE4; i += 256) {
        float4 v = b4[i];
        int i0 = i * 4;
        int ch = g * 32 + (i0 >> 12);             // 4096 pixels per channel
        int nn = i0 & (N_TOK - 1);                // 4 consecutive pixels, same ch
        float w = gw[ch] * rstd, bb = gb[ch] - mean * w;
        half_t* t = tokens + ((bi << 12) + nn) * C_DIM + ch;
        t[0]         = (half_t)(v.x * w + bb);
        t[C_DIM]     = (half_t)(v.y * w + bb);
        t[2 * C_DIM] = (half_t)(v.z * w + bb);
        t[3 * C_DIM] = (half_t)(v.w * w + bb);
    }
}

// ---------------------------------------------------------------- QKV GEMM (M=16384,N=768,K=256)
// Double-buffered async-to-LDS staging; C = tokens * W^T + b.
#define LDA 40   // halves, 80B rows (16B aligned)
__global__ __launch_bounds__(256) void qkv_gemm(const half_t* __restrict__ tokens,
                                                const half_t* __restrict__ wq,
                                                const float* __restrict__ qkv_b,
                                                half_t* __restrict__ Q,
                                                half_t* __restrict__ K,
                                                half_t* __restrict__ Vt) {
    __shared__ __align__(16) half_t As[2][128 * LDA];
    __shared__ __align__(16) half_t Bs[2][64 * LDA];
    const int tid = threadIdx.x;
    const int wave = tid >> 5, lane = tid & 31;
    const int laneLo = lane & 15, hi = lane >> 4;
    const int mBase = blockIdx.x * 128;
    const int oBase = blockIdx.y * 64;

    auto issue = [&](int kc, int buf) {
#pragma unroll
        for (int it = 0; it < 2; ++it) {              // A tile 128x32: 512 chunks
            int c = it * 256 + tid;
            int row = c >> 2, seg = (c & 3) * 8;
            async_ld16(&As[buf][row * LDA + seg], tokens,
                       (unsigned)(((mBase + row) * C_DIM + kc * 32 + seg) * 2));
        }
        {                                             // B tile 64x32: 256 chunks
            int row = tid >> 2, seg = (tid & 3) * 8;
            async_ld16(&Bs[buf][row * LDA + seg], wq,
                       (unsigned)(((oBase + row) * C_DIM + kc * 32 + seg) * 2));
        }
    };

    v8f acc[4]; { v8f z = {}; for (int t = 0; t < 4; ++t) acc[t] = z; }

    issue(0, 0);
    ASYNC_WAIT();
    __syncthreads();
    for (int kc = 0; kc < 8; ++kc) {
        int cur = kc & 1;
        if (kc < 7) issue(kc + 1, 1 - cur);
        // load all fragments into distinct registers, then WMMA back-to-back
        v16h a = ld_frag_a(&As[cur][(wave * 16 + laneLo) * LDA] + (hi ? 16 : 0));
        v16h bf[4];
#pragma unroll
        for (int t = 0; t < 4; ++t)
            bf[t] = ld_frag_b(&Bs[cur][(t * 16 + laneLo) * LDA] + (hi ? 16 : 0));
#pragma unroll
        for (int t = 0; t < 4; ++t)
            acc[t] = wmma_f16(a, bf[t], acc[t]);
        ASYNC_WAIT();
        __syncthreads();
    }

    // Q gets 1/sqrt(c) * log2(e) folded in so the flash softmax can use exp2.
    const float qscale = 0.0625f * LOG2E;
#pragma unroll
    for (int t = 0; t < 4; ++t) {
        int o = oBase + t * 16 + laneLo;
        float bias = qkv_b[o];
#pragma unroll
        for (int r = 0; r < 8; ++r) {
            int m = mBase + wave * 16 + r + 8 * hi;
            int bi = m >> 12, nn = m & (N_TOK - 1);
            float v = acc[t][r] + bias;
            if (o < 256)      Q[((bi << 12) + nn) * C_DIM + o] = (half_t)(v * qscale);
            else if (o < 512) K[((bi << 12) + nn) * C_DIM + (o - 256)] = (half_t)v;
            else              Vt[(bi * C_DIM + (o - 512)) * N_TOK + nn] = (half_t)v;
        }
    }
}

// ---------------------------------------------------------------- Flash attention
// grid (64, 4): block = 64 query rows, 4 waves x 16-row tile; Bc = 32 keys/iter.
// Double-buffered async K/V tiles; softmax in exp2 domain (scale folded into Q).
#define LDK 264  // halves (528B rows)
#define LDV 40   // halves (80B rows)
__global__ __launch_bounds__(128) void attn_kernel(const half_t* __restrict__ Q,
                                                   const half_t* __restrict__ Kg,
                                                   const half_t* __restrict__ Vt,
                                                   half_t* __restrict__ Ao) {
    __shared__ __align__(16) half_t Kt[2][32 * LDK];
    __shared__ __align__(16) half_t Vl[2][256 * LDV];
    __shared__ __align__(16) half_t Pl[4 * 16 * 32];

    const int tid = threadIdx.x;
    const int wave = tid >> 5, lane = tid & 31;
    const int laneLo = lane & 15, hi = lane >> 4;
    const int bi = blockIdx.y;
    const int mBase = blockIdx.x * 64;
    const int koff = hi ? 16 : 0;

    auto issue = [&](int j, int buf) {
#pragma unroll
        for (int it = 0; it < 8; ++it) {              // K tile: 1024 chunks
            int c = it * 128 + tid;
            int row = c >> 5, seg = (c & 31) * 8;
            async_ld16(&Kt[buf][row * LDK + seg], Kg,
                       (unsigned)((((bi << 12) + j * 32 + row) * C_DIM + seg) * 2));
        }
#pragma unroll
        for (int it = 0; it < 8; ++it) {              // V tile: 1024 chunks
            int c = it * 128 + tid;
            int row = c >> 2, seg = (c & 3) * 8;
            async_ld16(&Vl[buf][row * LDV + seg], Vt,
                       (unsigned)(((bi * C_DIM + row) * N_TOK + j * 32 + seg) * 2));
        }
    };

    // Q fragments resident in registers (A layout), 8 K-chunks of 32
    v16h qf[8];
    {
        const half_t* qrow = Q + ((bi << 12) + mBase + wave * 16 + laneLo) * C_DIM;
#pragma unroll
        for (int kc = 0; kc < 8; ++kc)
            qf[kc] = ld_frag_a(qrow + kc * 32 + koff);
    }

    v8f of_[16]; { v8f z = {}; for (int i = 0; i < 16; ++i) of_[i] = z; }
    float mrow[8], lrow[8];
#pragma unroll
    for (int r = 0; r < 8; ++r) { mrow[r] = -1e30f; lrow[r] = 0.f; }

    half_t* Pw = Pl + wave * 512;

    issue(0, 0);
    ASYNC_WAIT();
    __syncthreads();
    for (int j = 0; j < N_TOK / 32; ++j) {
        int cur = j & 1;
        if (j < N_TOK / 32 - 1) issue(j + 1, 1 - cur);

        // S = Q * K^T (16x32 scores, two C tiles), software-pipelined frag loads
        v8f s0 = {}, s1 = {};
        {
            v16h b0 = ld_frag_b(&Kt[cur][laneLo * LDK] + (hi ? 16 : 0));
            v16h b1 = ld_frag_b(&Kt[cur][(16 + laneLo) * LDK] + (hi ? 16 : 0));
#pragma unroll
            for (int kc = 0; kc < 8; ++kc) {
                v16h nb0 = b0, nb1 = b1;
                if (kc < 7) {
                    nb0 = ld_frag_b(&Kt[cur][laneLo * LDK + (kc + 1) * 32] + (hi ? 16 : 0));
                    nb1 = ld_frag_b(&Kt[cur][(16 + laneLo) * LDK + (kc + 1) * 32] + (hi ? 16 : 0));
                }
                s0 = wmma_f16(qf[kc], b0, s0);
                s1 = wmma_f16(qf[kc], b1, s1);
                b0 = nb0; b1 = nb1;
            }
        }

        // online softmax, exp2 domain (row = r + 8*hi per lane; cols across 16-lane halves)
#pragma unroll
        for (int r = 0; r < 8; ++r) {
            float a0 = s0[r], a1 = s1[r];
            float mx = fmaxf(a0, a1);
            for (int d = 1; d < 16; d <<= 1) mx = fmaxf(mx, __shfl_xor(mx, d));
            float mnew  = fmaxf(mrow[r], mx);
            float alpha = exp2f(mrow[r] - mnew);
            mrow[r] = mnew;
            float p0 = exp2f(a0 - mnew), p1 = exp2f(a1 - mnew);
            float sum = p0 + p1;
            for (int d = 1; d < 16; d <<= 1) sum += __shfl_xor(sum, d);
            lrow[r] = lrow[r] * alpha + sum;
            int m = r + 8 * hi;
            Pw[m * 32 + laneLo]      = (half_t)p0;
            Pw[m * 32 + 16 + laneLo] = (half_t)p1;
#pragma unroll
            for (int ct = 0; ct < 16; ++ct) of_[ct][r] *= alpha;
        }
        // P is wave-private; same-wave LDS ops are in-order -> compiler fence only
        WAVE_LDS_FENCE();

        // O += P * V (K = 32 keys, 16 channel tiles), software-pipelined frag loads
        {
            v16h pf = ld_frag_a(Pw + laneLo * 32 + koff);
            v16h bv = ld_frag_b(&Vl[cur][laneLo * LDV] + (hi ? 16 : 0));
#pragma unroll
            for (int ct = 0; ct < 16; ++ct) {
                v16h nbv = bv;
                if (ct < 15)
                    nbv = ld_frag_b(&Vl[cur][((ct + 1) * 16 + laneLo) * LDV] + (hi ? 16 : 0));
                of_[ct] = wmma_f16(pf, bv, of_[ct]);
                bv = nbv;
            }
        }

        ASYNC_WAIT();       // prefetch for j+1 landed
        __syncthreads();    // all waves done reading buf[cur] & writing next buf
    }

    // normalize + store f16 attention output [b,n,c]
#pragma unroll
    for (int r = 0; r < 8; ++r) {
        float invl = 1.0f / lrow[r];
        int m = mBase + wave * 16 + r + 8 * hi;
        half_t* dst = Ao + ((bi << 12) + m) * C_DIM + laneLo;
#pragma unroll
        for (int ct = 0; ct < 16; ++ct)
            dst[ct * 16] = (half_t)(of_[ct][r] * invl);
    }
}

// ---------------------------------------------------------------- proj + bias + residual (f32 out)
__global__ __launch_bounds__(256) void proj_gemm(const half_t* __restrict__ Ao,
                                                 const half_t* __restrict__ pw,
                                                 const float* __restrict__ pb,
                                                 const float* __restrict__ x,
                                                 float* __restrict__ out) {
    __shared__ __align__(16) half_t As[2][128 * LDA];
    __shared__ __align__(16) half_t Bs[2][64 * LDA];
    const int tid = threadIdx.x;
    const int wave = tid >> 5, lane = tid & 31;
    const int laneLo = lane & 15, hi = lane >> 4;
    const int mBase = blockIdx.x * 128;
    const int oBase = blockIdx.y * 64;

    auto issue = [&](int kc, int buf) {
#pragma unroll
        for (int it = 0; it < 2; ++it) {
            int c = it * 256 + tid;
            int row = c >> 2, seg = (c & 3) * 8;
            async_ld16(&As[buf][row * LDA + seg], Ao,
                       (unsigned)(((mBase + row) * C_DIM + kc * 32 + seg) * 2));
        }
        {
            int row = tid >> 2, seg = (tid & 3) * 8;
            async_ld16(&Bs[buf][row * LDA + seg], pw,
                       (unsigned)(((oBase + row) * C_DIM + kc * 32 + seg) * 2));
        }
    };

    v8f acc[4]; { v8f z = {}; for (int t = 0; t < 4; ++t) acc[t] = z; }

    issue(0, 0);
    ASYNC_WAIT();
    __syncthreads();
    for (int kc = 0; kc < 8; ++kc) {
        int cur = kc & 1;
        if (kc < 7) issue(kc + 1, 1 - cur);
        v16h a = ld_frag_a(&As[cur][(wave * 16 + laneLo) * LDA] + (hi ? 16 : 0));
        v16h bf[4];
#pragma unroll
        for (int t = 0; t < 4; ++t)
            bf[t] = ld_frag_b(&Bs[cur][(t * 16 + laneLo) * LDA] + (hi ? 16 : 0));
#pragma unroll
        for (int t = 0; t < 4; ++t)
            acc[t] = wmma_f16(a, bf[t], acc[t]);
        ASYNC_WAIT();
        __syncthreads();
    }

#pragma unroll
    for (int t = 0; t < 4; ++t) {
        int o = oBase + t * 16 + laneLo;
        float bias = pb[o];
#pragma unroll
        for (int r = 0; r < 8; ++r) {
            int m = mBase + wave * 16 + r + 8 * hi;
            int bi = m >> 12, nn = m & (N_TOK - 1);
            int idx = (bi * C_DIM + o) * N_TOK + nn;
            out[idx] = acc[t][r] + bias + x[idx];
        }
    }
}

// ----------------------------------------------------------------
extern "C" void kernel_launch(void* const* d_in, const int* in_sizes, int n_in,
                              void* d_out, int out_size, void* d_ws, size_t ws_size,
                              hipStream_t stream) {
    const float* x      = (const float*)d_in[0];
    const float* norm_w = (const float*)d_in[1];
    const float* norm_b = (const float*)d_in[2];
    const float* qkv_w  = (const float*)d_in[3];
    const float* qkv_b  = (const float*)d_in[4];
    const float* proj_w = (const float*)d_in[5];
    const float* proj_b = (const float*)d_in[6];
    float* out = (float*)d_out;

    char* ws = (char*)d_ws;
    const size_t TOKB = (size_t)B_DIM * N_TOK * C_DIM * sizeof(half_t); // 8 MiB
    half_t* tokens  = (half_t*)(ws);
    half_t* Qb      = (half_t*)(ws + TOKB);
    half_t* Kb      = (half_t*)(ws + 2 * TOKB);
    half_t* Vtb     = (half_t*)(ws + 3 * TOKB);
    half_t* Aob     = (half_t*)(ws + 4 * TOKB);
    half_t* qkvw_h  = (half_t*)(ws + 5 * TOKB);
    half_t* projw_h = (half_t*)(ws + 5 * TOKB + (size_t)3 * C_DIM * C_DIM * sizeof(half_t));

    cvt_weights<<<dim3(768), dim3(256), 0, stream>>>(qkv_w, proj_w, qkvw_h, projw_h);
    gn_kernel<<<dim3(B_DIM * 8), dim3(256), 0, stream>>>(x, norm_w, norm_b, tokens);
    qkv_gemm<<<dim3(128, 12), dim3(256), 0, stream>>>(tokens, qkvw_h, qkv_b, Qb, Kb, Vtb);
    attn_kernel<<<dim3(64, B_DIM), dim3(128), 0, stream>>>(Qb, Kb, Vtb, Aob);
    proj_gemm<<<dim3(128, 4), dim3(256), 0, stream>>>(Aob, projw_h, proj_b, x, out);
}
